// Multi_Task_Learning_1_22239340659252
// MI455X (gfx1250) — compile-verified, matrix operand order checked by reading, not run on hardware
//
#include <hip/hip_runtime.h>
#include <hip/hip_bf16.h>
#include <cmath>

typedef _Float16 v8h  __attribute__((ext_vector_type(8)));
typedef _Float16 v16h __attribute__((ext_vector_type(16)));
typedef float    v8f  __attribute__((ext_vector_type(8)));

#define TM 128
#define TN 128
#define TK 32
#define LSTR 40   // padded LDS row stride in f16 elems (80 bytes; 16B aligned, conflict-free)

enum { F16OUT = 1, TRANSSTORE = 2, ACCUM = 4 };

// ---------------------------------------------------------------------------
// NT GEMM: C[M,N] = alpha * A[M,K] * Bt[N,K]^T (+ bias[n]) ; fp32 accumulate
// via V_WMMA_F32_16X16X32_F16. Batched through element strides sA/sB/sC.
// Double-buffered LDS pipeline, ONE barrier per k-step (ping-pong buffers make
// the write-after-read barrier redundant). Next tile's global loads issue
// before the compute phase so HBM latency hides under WMMA.
// GUARD=false fast path has zero divergent branches.
// ---------------------------------------------------------------------------
template <bool GUARD>
__global__ __launch_bounds__(256)
void gemm_nt_f16(const _Float16* __restrict__ A, int lda, long long sA,
                 const _Float16* __restrict__ Bt, int ldb, long long sB,
                 const float* __restrict__ bias,
                 void* __restrict__ Cp, int ldc, long long sC,
                 int M, int N, int K, float alpha, int flags)
{
    __shared__ _Float16 As[2][TM * LSTR];
    __shared__ _Float16 Bs[2][TN * LSTR];

    const int tid  = threadIdx.x;
    const int lane = tid & 31;
    const int wid  = tid >> 5;
    const int wN   = (wid & 3) * 32;   // 4 waves across N  (4*32 = 128)
    const int wM   = (wid >> 2) * 64;  // 2 waves across M  (2*64 = 128)

    const int m0 = blockIdx.y * TM;
    const int n0 = blockIdx.x * TN;
    const int b  = blockIdx.z;

    const _Float16* Ab = A  + (long long)b * sA;
    const _Float16* Bb = Bt + (long long)b * sB;

    // --- per-thread staging slots: rows (tid>>2) and (tid>>2)+64, chunk (tid&3)
    const int rS = tid >> 2;           // 0..63
    const int kc = (tid & 3) * 8;      // 0,8,16,24
    const bool vA0 = !GUARD || (m0 + rS)      < M;
    const bool vA1 = !GUARD || (m0 + rS + 64) < M;
    const bool vB0 = !GUARD || (n0 + rS)      < N;
    const bool vB1 = !GUARD || (n0 + rS + 64) < N;

    const _Float16* pA0 = Ab + (long long)(m0 + rS)      * lda + kc;
    const _Float16* pA1 = Ab + (long long)(m0 + rS + 64) * lda + kc;
    const _Float16* pB0 = Bb + (long long)(n0 + rS)      * ldb + kc;
    const _Float16* pB1 = Bb + (long long)(n0 + rS + 64) * ldb + kc;

    auto g8 = [](const _Float16* p, bool valid) -> v8h {
        if (GUARD && !valid) { v8h z = {}; return z; }
        return *(const v8h*)p;
    };

    v8f acc[4][2] = {};

    const int fr = lane & 15;   // fragment row (M for A / N for Bt)
    const int kg = lane >> 4;   // K-group select per documented layout

    // --- prologue: stage tile 0 into registers
    v8h a0 = g8(pA0, vA0), a1 = g8(pA1, vA1);
    v8h b0 = g8(pB0, vB0), b1 = g8(pB1, vB1);

    const int nk = K / TK;
    for (int i = 0; i < nk; ++i) {
        const int buf = i & 1;
        // commit staged registers to LDS
        *(v8h*)(As[buf] + rS * LSTR + kc)        = a0;
        *(v8h*)(As[buf] + (rS + 64) * LSTR + kc) = a1;
        *(v8h*)(Bs[buf] + rS * LSTR + kc)        = b0;
        *(v8h*)(Bs[buf] + (rS + 64) * LSTR + kc) = b1;
        __syncthreads();   // single barrier per step; ping-pong covers WAR

        // prefetch next k-tile (latency hides under the WMMA phase below)
        if (i + 1 < nk) {
            pA0 += TK; pA1 += TK; pB0 += TK; pB1 += TK;
            a0 = g8(pA0, vA0); a1 = g8(pA1, vA1);
            b0 = g8(pB0, vB0); b1 = g8(pB1, vB1);
        }

        const _Float16* __restrict__ Ac = As[buf];
        const _Float16* __restrict__ Bc = Bs[buf];

        // B fragments (2 x 16x32 tiles, loaded transposed-symmetric to A)
        v16h bf[2];
#pragma unroll
        for (int tn = 0; tn < 2; ++tn) {
            const _Float16* bp = Bc + (wN + tn * 16 + fr) * LSTR + kg * 8;
            v8h lo = *(const v8h*)bp;
            v8h hi = *(const v8h*)(bp + 16);
            bf[tn] = __builtin_shufflevector(lo, hi, 0,1,2,3,4,5,6,7,8,9,10,11,12,13,14,15);
        }
        // A fragments (4 x 16x32 tiles) + 8 WMMA per k-step per wave
#pragma unroll
        for (int tm = 0; tm < 4; ++tm) {
            const _Float16* ap = Ac + (wM + tm * 16 + fr) * LSTR + kg * 8;
            v8h lo = *(const v8h*)ap;
            v8h hi = *(const v8h*)(ap + 16);
            v16h af = __builtin_shufflevector(lo, hi, 0,1,2,3,4,5,6,7,8,9,10,11,12,13,14,15);
#pragma unroll
            for (int tn = 0; tn < 2; ++tn)
                acc[tm][tn] = __builtin_amdgcn_wmma_f32_16x16x32_f16(
                    false, af, false, bf[tn], (short)0, acc[tm][tn], false, false);
        }
    }

    // Epilogue: D layout -> VGPR r holds (M = r + 8*(lane>>4), N = lane&15)
    const int mE = (lane >> 4) * 8;
    const int nE = lane & 15;
#pragma unroll
    for (int tm = 0; tm < 4; ++tm)
#pragma unroll
        for (int tn = 0; tn < 2; ++tn)
#pragma unroll
            for (int rr = 0; rr < 8; ++rr) {
                int gm = m0 + wM + tm * 16 + mE + rr;
                int gn = n0 + wN + tn * 16 + nE;
                if (!GUARD || (gm < M && gn < N)) {
                    float v = acc[tm][tn][rr] * alpha;
                    if (bias) v += bias[gn];
                    long long idx = (flags & TRANSSTORE) ? ((long long)gn * ldc + gm)
                                                         : ((long long)gm * ldc + gn);
                    idx += (long long)b * sC;
                    if (flags & F16OUT) {
                        ((_Float16*)Cp)[idx] = (_Float16)v;
                    } else {
                        float* Cf = (float*)Cp;
                        if (flags & ACCUM) Cf[idx] += v; else Cf[idx] = v;
                    }
                }
            }
}

// ---------------------------------------------------------------------------
// In-place row softmax over an f16 matrix; one block (256 threads) per row.
// cols must be <= 16*256 (we have 768 / 2304).
// ---------------------------------------------------------------------------
__global__ __launch_bounds__(256)
void softmax_rows_f16(_Float16* __restrict__ S, int cols)
{
    __shared__ float red[256];
    long long row = blockIdx.x;
    _Float16* p = S + row * (long long)cols;

    float v[16];
    int cnt = 0;
    float m = -3.0e38f;
    for (int i = threadIdx.x; i < cols; i += 256) {
        float x = (float)p[i];
        v[cnt++] = x;
        m = fmaxf(m, x);
    }
    red[threadIdx.x] = m;
    __syncthreads();
    for (int s = 128; s > 0; s >>= 1) {
        if (threadIdx.x < s) red[threadIdx.x] = fmaxf(red[threadIdx.x], red[threadIdx.x + s]);
        __syncthreads();
    }
    m = red[0];
    __syncthreads();

    float sum = 0.0f;
    for (int j = 0; j < cnt; ++j) { v[j] = __expf(v[j] - m); sum += v[j]; }
    red[threadIdx.x] = sum;
    __syncthreads();
    for (int s = 128; s > 0; s >>= 1) {
        if (threadIdx.x < s) red[threadIdx.x] += red[threadIdx.x + s];
        __syncthreads();
    }
    float inv = 1.0f / red[0];

    cnt = 0;
    for (int i = threadIdx.x; i < cols; i += 256) p[i] = (_Float16)(v[cnt++] * inv);
}

// ---------------------------------------------------------------------------
// Tiled transpose with dtype convert: dst[c, r] = (TO)src[r, c]
// src is [srcRows, srcCols] ld srcLd; dst is [srcCols, srcRows] ld dstLd.
// ---------------------------------------------------------------------------
template <typename TI, typename TO>
__global__ __launch_bounds__(256)
void transpose_conv(const TI* __restrict__ src, long long srcLd,
                    TO* __restrict__ dst, long long dstLd,
                    int srcRows, int srcCols)
{
    __shared__ float tile[32][33];
    const int bx = blockIdx.x * 32;   // src col block
    const int by = blockIdx.y * 32;   // src row block
    const int tx = threadIdx.x & 31;
    const int ty = threadIdx.x >> 5;  // 0..7
#pragma unroll
    for (int j = 0; j < 32; j += 8) {
        int rIn = by + ty + j, cIn = bx + tx;
        tile[ty + j][tx] = (rIn < srcRows && cIn < srcCols)
                               ? (float)src[(long long)rIn * srcLd + cIn] : 0.0f;
    }
    __syncthreads();
#pragma unroll
    for (int j = 0; j < 32; j += 8) {
        int rOut = bx + ty + j;  // src col
        int cOut = by + tx;      // src row
        if (rOut < srcCols && cOut < srcRows)
            dst[(long long)rOut * dstLd + cOut] = (TO)tile[tx][ty + j];
    }
}

__global__ __launch_bounds__(256)
void convert_f32_f16(const float* __restrict__ src, _Float16* __restrict__ dst, long long n)
{
    long long i = (long long)blockIdx.x * 256 + threadIdx.x;
    long long stride = (long long)gridDim.x * 256;
    for (; i < n; i += stride) dst[i] = (_Float16)src[i];
}

// ---------------------------------------------------------------------------
extern "C" void kernel_launch(void* const* d_in, const int* in_sizes, int n_in,
                              void* d_out, int out_size, void* d_ws, size_t ws_size,
                              hipStream_t stream)
{
    (void)in_sizes; (void)n_in; (void)out_size; (void)ws_size;

    const int C  = 768, N = 2304, C3 = 3 * C, N3 = 3 * N;
    const int HN = 8, HD = 96, HDC = 288;
    const float scale   = 1.0f / sqrtf((float)HD);
    const float scale_c = 1.0f / sqrtf((float)HDC);

    const float* x    = (const float*)d_in[0];
    const float* y    = (const float*)d_in[1];
    const float* Wqx  = (const float*)d_in[2];
    const float* bqx  = (const float*)d_in[3];
    const float* Wqy  = (const float*)d_in[4];
    const float* bqy  = (const float*)d_in[5];
    const float* Wqxc = (const float*)d_in[6];
    const float* bqxc = (const float*)d_in[7];
    const float* Wqyc = (const float*)d_in[8];
    const float* bqyc = (const float*)d_in[9];
    const float* Wpxc = (const float*)d_in[10];
    const float* bpxc = (const float*)d_in[11];
    const float* Wpyc = (const float*)d_in[12];
    const float* bpyc = (const float*)d_in[13];
    const float* Wax  = (const float*)d_in[14];
    const float* bax  = (const float*)d_in[15];
    const float* Way  = (const float*)d_in[16];
    const float* bay  = (const float*)d_in[17];

    float* outX = (float*)d_out;
    float* outY = outX + (size_t)C * N;

    // --- carve workspace (f16 buffers, 256B aligned) ---
    char* wsBase = (char*)d_ws;
    size_t off = 0;
    auto allocH = [&](size_t elems) -> _Float16* {
        _Float16* p = (_Float16*)(wsBase + off);
        off += ((elems * sizeof(_Float16)) + 255) & ~(size_t)255;
        return p;
    };
    _Float16* xT     = allocH((size_t)N * C);     // [N,C]
    _Float16* yT     = allocH((size_t)N * C);
    _Float16* xc     = allocH((size_t)C * N);     // [C,N]
    _Float16* yc     = allocH((size_t)C * N);
    _Float16* WqT    = allocH((size_t)C3 * C);    // shared: Wqx^T then Wqy^T
    _Float16* WqcT   = allocH((size_t)N3 * N);    // shared: Wqxc^T then Wqyc^T
    _Float16* WpT    = allocH((size_t)N * N);     // shared: Wpxc^T then Wpyc^T
    _Float16* WaT    = allocH((size_t)C * C);     // shared: Wax^T then Way^T
    _Float16* qkvX   = allocH((size_t)N * C3);    // [N,3C]
    _Float16* qkvY   = allocH((size_t)N * C3);
    _Float16* qkvXc  = allocH((size_t)C * N3);    // [C,3N]
    _Float16* qkvYc  = allocH((size_t)C * N3);
    _Float16* vxT    = allocH((size_t)C * N);     // [C,N]
    _Float16* vyT    = allocH((size_t)C * N);
    _Float16* vxcT   = allocH((size_t)N * C);     // [N,C]
    _Float16* vycT   = allocH((size_t)N * C);
    _Float16* Sbuf   = allocH((size_t)HN * N * N);   // spatial scores (reused x/y)
    _Float16* Cbuf   = allocH((size_t)HN * C * C);   // channel scores (reused x/y)
    _Float16* attnS  = allocH((size_t)N * C);        // [8,N,hd] == [N,C] flat
    _Float16* attnC  = allocH((size_t)C * N);        // [8,C,hdc] == [C,N] flat

    auto cdiv = [](int a, int b) { return (a + b - 1) / b; };

    auto gemm = [&](const _Float16* A, int lda, long long sA,
                    const _Float16* Bt, int ldb, long long sB,
                    const float* bias, void* Cp, int ldc, long long sC,
                    int M, int Nn, int K, float alpha, int flags, int batch) {
        dim3 grid(cdiv(Nn, TN), cdiv(M, TM), batch);
        if ((M % TM) || (Nn % TN))
            gemm_nt_f16<true><<<grid, 256, 0, stream>>>(A, lda, sA, Bt, ldb, sB, bias,
                                                        Cp, ldc, sC, M, Nn, K, alpha, flags);
        else
            gemm_nt_f16<false><<<grid, 256, 0, stream>>>(A, lda, sA, Bt, ldb, sB, bias,
                                                         Cp, ldc, sC, M, Nn, K, alpha, flags);
    };
    auto trF = [&](const float* src, long long srcLd, _Float16* dst, long long dstLd,
                   int rows, int cols) {
        transpose_conv<float, _Float16><<<dim3(cdiv(cols, 32), cdiv(rows, 32)), 256, 0, stream>>>(
            src, srcLd, dst, dstLd, rows, cols);
    };
    auto trH = [&](const _Float16* src, long long srcLd, _Float16* dst, long long dstLd,
                   int rows, int cols) {
        transpose_conv<_Float16, _Float16><<<dim3(cdiv(cols, 32), cdiv(rows, 32)), 256, 0, stream>>>(
            src, srcLd, dst, dstLd, rows, cols);
    };
    auto sm = [&](_Float16* S, int rows, int cols) {
        softmax_rows_f16<<<rows, 256, 0, stream>>>(S, cols);
    };

    // --- input prep ---
    trF(x, N, xT, C, C, N);                       // x [C,N] -> xT [N,C]
    trF(y, N, yT, C, C, N);
    convert_f32_f16<<<cdiv(C * N, 256), 256, 0, stream>>>(x, xc, (long long)C * N);
    convert_f32_f16<<<cdiv(C * N, 256), 256, 0, stream>>>(y, yc, (long long)C * N);

    // --- qkv projections (spatial + channel), bias fused ---
    trF(Wqx, C3, WqT, C, C, C3);                  // [C,3C] -> [3C,C]
    gemm(xT, C, 0, WqT, C, 0, bqx, qkvX, C3, 0, N, C3, C, 1.0f, F16OUT, 1);
    trF(Wqy, C3, WqT, C, C, C3);
    gemm(yT, C, 0, WqT, C, 0, bqy, qkvY, C3, 0, N, C3, C, 1.0f, F16OUT, 1);

    trF(Wqxc, N3, WqcT, N, N, N3);                // [N,3N] -> [3N,N]
    gemm(xc, N, 0, WqcT, N, 0, bqxc, qkvXc, N3, 0, C, N3, N, 1.0f, F16OUT, 1);
    trF(Wqyc, N3, WqcT, N, N, N3);
    gemm(yc, N, 0, WqcT, N, 0, bqyc, qkvYc, N3, 0, C, N3, N, 1.0f, F16OUT, 1);

    // --- V transposes (for NT AV gemms) ---
    trH(qkvX + 2 * C, C3, vxT, N, N, C);          // v_x [N,C] -> [C,N]
    trH(qkvY + 2 * C, C3, vyT, N, N, C);
    trH(qkvXc + 2 * N, N3, vxcT, C, C, N);        // v_xc [C,N] -> [N,C]
    trH(qkvYc + 2 * N, N3, vycT, C, C, N);

    const long long sNN = (long long)N * N, sCC = (long long)C * C;

    // --- spatial self-attention, x ---
    gemm(qkvX, C3, HD, qkvX + C, C3, HD, nullptr, Sbuf, N, sNN,
         N, N, HD, scale, F16OUT, HN);
    sm(Sbuf, HN * N, N);
    gemm(Sbuf, N, sNN, vxT, N, (long long)HD * N, nullptr, attnS, HD, (long long)N * HD,
         N, HD, N, 1.0f, F16OUT, HN);
    trF(Wax, C, WaT, C, C, C);
    gemm(attnS, C, 0, WaT, C, 0, bax, outX, N, 0, N, C, C, 1.0f, TRANSSTORE, 1);

    // --- spatial self-attention, y (reuse Sbuf/attnS/WaT) ---
    gemm(qkvY, C3, HD, qkvY + C, C3, HD, nullptr, Sbuf, N, sNN,
         N, N, HD, scale, F16OUT, HN);
    sm(Sbuf, HN * N, N);
    gemm(Sbuf, N, sNN, vyT, N, (long long)HD * N, nullptr, attnS, HD, (long long)N * HD,
         N, HD, N, 1.0f, F16OUT, HN);
    trF(Way, C, WaT, C, C, C);
    gemm(attnS, C, 0, WaT, C, 0, bay, outY, N, 0, N, C, C, 1.0f, TRANSSTORE, 1);

    // --- channel cross-attention, x queries -> y keys/values ---
    gemm(qkvXc, N3, HDC, qkvYc + N, N3, HDC, nullptr, Cbuf, C, sCC,
         C, C, HDC, scale_c, F16OUT, HN);
    sm(Cbuf, HN * C, C);
    gemm(Cbuf, C, sCC, vycT, C, (long long)HDC * C, nullptr, attnC, HDC, (long long)C * HDC,
         C, HDC, C, 1.0f, F16OUT, HN);
    trF(Wpxc, N, WpT, N, N, N);
    gemm(attnC, N, 0, WpT, N, 0, bpxc, outX, N, 0, C, N, N, 1.0f, ACCUM, 1);

    // --- channel cross-attention, y queries -> x keys/values (reuse Cbuf/attnC/WpT) ---
    gemm(qkvYc, N3, HDC, qkvXc + N, N3, HDC, nullptr, Cbuf, C, sCC,
         C, C, HDC, scale_c, F16OUT, HN);
    sm(Cbuf, HN * C, C);
    gemm(Cbuf, C, sCC, vxcT, C, (long long)HDC * C, nullptr, attnC, HDC, (long long)C * HDC,
         C, HDC, C, 1.0f, F16OUT, HN);
    trF(Wpyc, N, WpT, N, N, N);
    gemm(attnC, N, 0, WpT, N, 0, bpyc, outY, N, 0, C, N, N, 1.0f, ACCUM, 1);
}
